// GNNStack_8581344657723
// MI455X (gfx1250) — compile-verified, matrix operand order checked by reading, not run on hardware
//
#include <hip/hip_runtime.h>
#include <hip/hip_bf16.h>

typedef __attribute__((ext_vector_type(16))) __bf16   v16bf;
typedef __attribute__((ext_vector_type(4)))  __bf16   v4bf;
typedef __attribute__((ext_vector_type(8)))  float    v8f;
typedef __attribute__((ext_vector_type(4)))  uint32_t u32x4;
typedef __attribute__((ext_vector_type(8)))  uint32_t u32x8;

#define D 128
#define WPITCH 136   // bf16 halfs per row in LDS: 272B = 68 dwords -> +4 banks/row

__device__ __forceinline__ float bf2f(unsigned short u) {
  union { unsigned u32; float f; } c; c.u32 = ((unsigned)u) << 16; return c.f;
}

// Hardware no-return f32 atomic add at device scope (L2-resident RMW).
__device__ __forceinline__ void atom_add_f32(float* p, float v) {
  asm volatile("global_atomic_add_f32 %0, %1, off scope:SCOPE_DEV"
               :: "v"(p), "v"(v) : "memory");
}

__device__ __forceinline__ void pack8(v16bf& dst, int base, const float4& a, const float4& b) {
  dst[base+0] = (__bf16)a.x; dst[base+1] = (__bf16)a.y;
  dst[base+2] = (__bf16)a.z; dst[base+3] = (__bf16)a.w;
  dst[base+4] = (__bf16)b.x; dst[base+5] = (__bf16)b.y;
  dst[base+6] = (__bf16)b.z; dst[base+7] = (__bf16)b.w;
}

// -----------------------------------------------------------------------------
// Fused GEMM (+bias, optional row-L2-normalize): H[N,128](bf16) = norm(X @ W^T + b)
// W is TDM-DMA'd into LDS (f32, 64KB), converted once per block to a padded bf16
// LDS image; B-fragments are then 2x ds_load_b128 per tile with no per-use cvt.
// One wave per 16-row strip; 8 tiles of 16x16; K=128 in 4 WMMA steps.
// -----------------------------------------------------------------------------
template<bool NORM>
__global__ __launch_bounds__(256)
void gemm_rows16(const float* __restrict__ X, const float* __restrict__ W,
                 const float* __restrict__ bias, __bf16* __restrict__ H, int N) {
  extern __shared__ char smem[];                 // [0,64K): W f32 ; [64K,..): W bf16 padded
  __bf16* wb = (__bf16*)(smem + 64 * 1024);

  // --- Stage 1: TDM load of W[128,128] f32 -> LDS offset 0 (wave 0 only; EXEC ignored)
  if ((threadIdx.x >> 5) == 0) {
    const uint64_t ga = (uint64_t)(uintptr_t)W;
    u32x4 g0 = { 1u,                                   // count=1, user mode
                 0u,                                   // lds_addr = 0 (dynamic LDS base)
                 (uint32_t)ga,
                 (uint32_t)((ga >> 32) & 0x1FFFFFFu) | (2u << 30) };  // addr[56:32] | type=2
    u32x8 g1 = { 2u << 16,        // workgroup_mask=0, data_size=2 (4B)
                 128u << 16,      // tensor_dim0 lo16 = 128
                 128u << 16,      // tensor_dim0 hi16=0 | tensor_dim1 lo16 = 128
                 128u << 16,      // tensor_dim1 hi16=0 | tile_dim0 = 128
                 128u,            // tile_dim1 = 128, tile_dim2 = 0
                 128u,            // tensor_dim0_stride lo32 = 128
                 0u, 0u };        // stride hi | tensor_dim1_stride = 0
    asm volatile("tensor_load_to_lds %0, %1" :: "s"(g0), "s"(g1) : "memory");
    __builtin_amdgcn_s_wait_tensorcnt(0);
  }
  __syncthreads();

  // --- Stage 2: one-shot f32 -> bf16 conversion into padded LDS image
  {
    const float4* wf4 = (const float4*)smem;
    const int r  = threadIdx.x >> 1;               // row 0..127
    const int h2 = threadIdx.x & 1;                // half-row
    #pragma unroll
    for (int j = 0; j < 16; ++j) {
      float4 v = wf4[(r * D + h2 * 64 + j * 4) >> 2];
      v4bf t = { (__bf16)v.x, (__bf16)v.y, (__bf16)v.z, (__bf16)v.w };
      *(v4bf*)(wb + r * WPITCH + h2 * 64 + j * 4) = t;
    }
  }
  __syncthreads();

  const int strips = (N + 15) >> 4;
  const int wave = blockIdx.x * (blockDim.x >> 5) + (threadIdx.x >> 5);
  if (wave >= strips) return;                      // wave-uniform: EXEC stays all-ones
  const int lane = threadIdx.x & 31;
  const int m16  = lane & 15;
  const int hi   = lane >> 4;

  int arowi = wave * 16 + m16; if (arowi >= N) arowi = N - 1;
  const float* arow = X + (size_t)arowi * D;

  v8f c[8] = {};

  #pragma unroll
  for (int kb = 0; kb < D; kb += 32) {
    v16bf a;
    const float4* ap0 = (const float4*)(arow + kb + hi * 8);
    const float4* ap1 = (const float4*)(arow + kb + 16 + hi * 8);
    pack8(a, 0, ap0[0], ap0[1]);
    pack8(a, 8, ap1[0], ap1[1]);
    #pragma unroll
    for (int t = 0; t < 8; ++t) {
      const int n = t * 16 + m16;                  // B col = W row (B[k,n] = W[n,k])
      union { uint4 q[2]; v16bf v; } bb;
      bb.q[0] = *(const uint4*)(wb + n * WPITCH + kb + hi * 8);
      bb.q[1] = *(const uint4*)(wb + n * WPITCH + kb + 16 + hi * 8);
      c[t] = __builtin_amdgcn_wmma_f32_16x16x32_bf16(
          false, a, false, bb.v, (short)0, c[t], false, false);
    }
  }

  #pragma unroll
  for (int t = 0; t < 8; ++t) {
    float bv = bias[t * 16 + m16];
    #pragma unroll
    for (int i = 0; i < 8; ++i) c[t][i] += bv;
  }

  if (NORM) {
    // row r = wave*16 + hi*8 + i lives at VGPR slot i across 16 lanes of a half
    #pragma unroll
    for (int i = 0; i < 8; ++i) {
      float s = 0.f;
      #pragma unroll
      for (int t = 0; t < 8; ++t) s += c[t][i] * c[t][i];
      s += __shfl_xor(s, 1, 32);
      s += __shfl_xor(s, 2, 32);
      s += __shfl_xor(s, 4, 32);
      s += __shfl_xor(s, 8, 32);
      float inv = 1.f / fmaxf(__builtin_sqrtf(s), 1e-12f);
      #pragma unroll
      for (int t = 0; t < 8; ++t) c[t][i] *= inv;
    }
  }

  #pragma unroll
  for (int i = 0; i < 8; ++i) {
    int r = wave * 16 + hi * 8 + i;
    if (r < N) {
      __bf16* hrow = H + (size_t)r * D + m16;
      #pragma unroll
      for (int t = 0; t < 8; ++t) hrow[t * 16] = (__bf16)c[t][i];
    }
  }
}

// -----------------------------------------------------------------------------
// Scatter-mean accumulate: one wave per edge, 4 dims per lane.
// agg (f32, 51.2MB -> L2-resident) += bf16 gather of h[src]; cnt[dst] += 1.
// -----------------------------------------------------------------------------
__global__ __launch_bounds__(256)
void scatter_mean_acc(const __bf16* __restrict__ H,
                      const long long* __restrict__ src,
                      const long long* __restrict__ dst,
                      float* __restrict__ agg, float* __restrict__ cnt, int E) {
  const int e = blockIdx.x * (blockDim.x >> 5) + (threadIdx.x >> 5);
  if (e >= E) return;
  const int lane = threadIdx.x & 31;
  const int s = (int)src[e];
  const int d = (int)dst[e];
  const ushort2* hp = (const ushort2*)((const unsigned short*)H + (size_t)s * D + lane * 4);
  ushort2 h0 = hp[0], h1 = hp[1];
  float* ap = agg + (size_t)d * D + lane * 4;
  atom_add_f32(ap + 0, bf2f(h0.x));
  atom_add_f32(ap + 1, bf2f(h0.y));
  atom_add_f32(ap + 2, bf2f(h1.x));
  atom_add_f32(ap + 3, bf2f(h1.y));
  if (lane == 0) atom_add_f32(cnt + d, 1.0f);
}

// -----------------------------------------------------------------------------
// x_next = relu(agg / max(cnt,1)) : 32 threads x float4 per row
// -----------------------------------------------------------------------------
__global__ __launch_bounds__(256)
void finalize_mean_relu(const float* __restrict__ agg, const float* __restrict__ cnt,
                        float* __restrict__ Xn, int N) {
  const int t = blockIdx.x * blockDim.x + threadIdx.x;
  const int row = t >> 5, l = t & 31;
  if (row >= N) return;
  const float inv = 1.0f / fmaxf(cnt[row], 1.0f);
  float4 v = *(const float4*)(agg + (size_t)row * D + l * 4);
  v.x = fmaxf(v.x * inv, 0.f); v.y = fmaxf(v.y * inv, 0.f);
  v.z = fmaxf(v.z * inv, 0.f); v.w = fmaxf(v.w * inv, 0.f);
  *(float4*)(Xn + (size_t)row * D + l * 4) = v;
}

// -----------------------------------------------------------------------------
// Final 128->40 projection + log_softmax. W2 (20KB) staged in LDS.
// 4 lanes per row; quad reduction via shfl_xor; each lane writes 10 outputs.
// -----------------------------------------------------------------------------
__global__ __launch_bounds__(256)
void mlp2_logsoftmax(const __bf16* __restrict__ H1, const float* __restrict__ W2,
                     const float* __restrict__ b2, float* __restrict__ out, int N) {
  __shared__ float w2s[40 * D];
  __shared__ float b2s[40];
  for (int i = threadIdx.x; i < 40 * D; i += blockDim.x) w2s[i] = W2[i];
  if (threadIdx.x < 40) b2s[threadIdx.x] = b2[threadIdx.x];
  __syncthreads();

  const int gt = blockIdx.x * blockDim.x + threadIdx.x;
  const int row = gt >> 2, q = gt & 3;
  if (row >= N) return;

  float r[32];
  const uint4* hp = (const uint4*)((const unsigned short*)H1 + (size_t)row * D + q * 32);
  #pragma unroll
  for (int j = 0; j < 4; ++j) {
    uint4 u = hp[j];
    unsigned w[4] = {u.x, u.y, u.z, u.w};
    #pragma unroll
    for (int k = 0; k < 4; ++k) {
      r[j * 8 + k * 2 + 0] = bf2f((unsigned short)(w[k] & 0xffffu));
      r[j * 8 + k * 2 + 1] = bf2f((unsigned short)(w[k] >> 16));
    }
  }

  float acc[40];
  #pragma unroll
  for (int o = 0; o < 40; ++o) {
    const float* wr = w2s + o * D + q * 32;
    float s = 0.f;
    #pragma unroll
    for (int i = 0; i < 32; ++i) s += r[i] * wr[i];
    acc[o] = s;
  }
  #pragma unroll
  for (int o = 0; o < 40; ++o) {
    acc[o] += __shfl_xor(acc[o], 1, 32);
    acc[o] += __shfl_xor(acc[o], 2, 32);
    acc[o] += b2s[o];
  }
  float m = acc[0];
  #pragma unroll
  for (int o = 1; o < 40; ++o) m = fmaxf(m, acc[o]);
  float s = 0.f;
  #pragma unroll
  for (int o = 0; o < 40; ++o) s += __expf(acc[o] - m);
  const float lse = m + __logf(s);
  #pragma unroll
  for (int j = 0; j < 10; ++j) {
    int o = q * 10 + j;
    out[(size_t)row * 40 + o] = acc[o] - lse;
  }
}

extern "C" void kernel_launch(void* const* d_in, const int* in_sizes, int n_in,
                              void* d_out, int out_size, void* d_ws, size_t ws_size,
                              hipStream_t stream) {
  const float*     x   = (const float*)d_in[0];
  const long long* ei  = (const long long*)d_in[1];   // int64 edge_index [2,E]
  const float* Wl[3] = {(const float*)d_in[2], (const float*)d_in[4], (const float*)d_in[6]};
  const float* bl[3] = {(const float*)d_in[3], (const float*)d_in[5], (const float*)d_in[7]};
  const float* Wp1 = (const float*)d_in[8];
  const float* bp1 = (const float*)d_in[9];
  const float* Wp2 = (const float*)d_in[10];
  const float* bp2 = (const float*)d_in[11];

  const int N = in_sizes[0] / D;
  const int E = in_sizes[1] / 2;
  const long long* srcp = ei;
  const long long* dstp = ei + E;

  char* ws = (char*)d_ws;
  __bf16* hB = (__bf16*)ws;            size_t o = (size_t)N * D * 2;   // 25.6 MB
  float*  agg = (float*)(ws + o);      o += (size_t)N * D * 4;         // 51.2 MB
  float*  xb  = (float*)(ws + o);      o += (size_t)N * D * 4;         // 51.2 MB
  float*  cnt = (float*)(ws + o);                                      // 0.4 MB

  const int strips = (N + 15) / 16;
  const size_t smem = 64 * 1024 + (size_t)D * WPITCH * 2;              // ~98 KB
  dim3 B(256);
  dim3 gG((strips + 7) / 8);
  dim3 gS((E + 7) / 8);
  dim3 gF((unsigned)(((size_t)N * 32 + 255) / 256));
  dim3 gM((unsigned)(((size_t)N * 4 + 255) / 256));

  const float* xin = x;
  for (int l = 0; l < 3; ++l) {
    gemm_rows16<true><<<gG, B, smem, stream>>>(xin, Wl[l], bl[l], hB, N);
    hipMemsetAsync(agg, 0, (size_t)N * D * 4, stream);
    hipMemsetAsync(cnt, 0, (size_t)N * 4, stream);
    scatter_mean_acc<<<gS, B, 0, stream>>>(hB, srcp, dstp, agg, cnt, E);
    finalize_mean_relu<<<gF, B, 0, stream>>>(agg, cnt, xb, N);
    xin = xb;
  }
  gemm_rows16<false><<<gG, B, smem, stream>>>(xin, Wp1, bp1, hB, N);
  mlp2_logsoftmax<<<gM, B, 0, stream>>>(hB, Wp2, bp2, (float*)d_out, N);
}
